// SupConLoss_39204461477965
// MI455X (gfx1250) — compile-verified, matrix operand order checked by reading, not run on hardware
//
#include <hip/hip_runtime.h>
#include <math.h>

#define NROWS 16384
#define DIM   128
#define INV_T 14.285714285714286f   // 1 / 0.07

typedef _Float16 v16h __attribute__((ext_vector_type(16)));
typedef _Float16 v8h  __attribute__((ext_vector_type(8)));
typedef _Float16 v4h  __attribute__((ext_vector_type(4)));
typedef float    v8f  __attribute__((ext_vector_type(8)));
typedef float    v4f  __attribute__((ext_vector_type(4)));

// ---------------------------------------------------------------------------
// Kernel 0: zero the P/A accumulators and the scalar output.
// ---------------------------------------------------------------------------
__global__ void zero_kernel(float* __restrict__ acc, float* __restrict__ out, int n) {
    int i = blockIdx.x * blockDim.x + threadIdx.x;
    if (i < n) acc[i] = 0.0f;
    if (i == 0) out[0] = 0.0f;
}

// ---------------------------------------------------------------------------
// Kernel 1: L2-normalize each row (eps=1e-12) and convert to f16.
// One wave32 per row; each lane handles 4 contiguous floats.
// ---------------------------------------------------------------------------
__global__ void normalize_kernel(const float* __restrict__ feats,
                                 _Float16* __restrict__ fh) {
    const int gwave = (blockIdx.x * blockDim.x + threadIdx.x) >> 5;
    const int lane  = threadIdx.x & 31;
    if (gwave >= NROWS) return;

    const float* rp = feats + (size_t)gwave * DIM + lane * 4;
    v4f x = *(const v4f*)rp;
    float s = x.x * x.x + x.y * x.y + x.z * x.z + x.w * x.w;
#pragma unroll
    for (int off = 16; off >= 1; off >>= 1) s += __shfl_xor(s, off, 32);

    const float scale = 1.0f / fmaxf(sqrtf(s), 1e-12f);
    v4h y;
    y.x = (_Float16)(x.x * scale);
    y.y = (_Float16)(x.y * scale);
    y.z = (_Float16)(x.z * scale);
    y.w = (_Float16)(x.w * scale);
    *(v4h*)(fh + (size_t)gwave * DIM + lane * 4) = y;
}

// ---------------------------------------------------------------------------
// WMMA 16x16x32 f16 A-fragment loader (B fragment of feats^T is identical).
// ISA 7.12.2: lane m (0-15) holds row m, K = {k0..k0+7, k0+16..k0+23};
// lane m+16 holds row m, K = {k0+8..k0+15, k0+24..k0+31}.
// Both chunks are contiguous 16-byte (8 x f16) aligned loads.
// ---------------------------------------------------------------------------
__device__ __forceinline__ v16h load_frag(const _Float16* __restrict__ base,
                                          int row, int k0, int hi) {
    union { v16h v; v8h h[2]; } u;
    const _Float16* p = base + (size_t)row * DIM + k0 + hi * 8;
    u.h[0] = *(const v8h*)p;
    u.h[1] = *(const v8h*)(p + 16);
    return u.v;
}

// ---------------------------------------------------------------------------
// Kernel 2: fused tiled Gram + exp + masked row-sums.
// Block = 256 threads (8 waves), tile = 128(i) x 128(j).
// Wave w owns the 16-row strip r0 = bx*128 + w*16 across all 8 j-subtiles.
// ---------------------------------------------------------------------------
__global__ void __launch_bounds__(256, 2)
supcon_tiles(const _Float16* __restrict__ fh,
             const int* __restrict__ labels,
             float* __restrict__ Psum,   // sum of exp over same-label, off-diag
             float* __restrict__ Asum) { // sum of exp over ALL j (incl. diag)
    const int lane = threadIdx.x & 31;
    const int wave = threadIdx.x >> 5;
    const int m    = lane & 15;
    const int hi   = lane >> 4;

    const int r0    = blockIdx.x * 128 + wave * 16;
    const int cbase = blockIdx.y * 128;

    // A fragments for this wave's 16 rows, all K (128 = 4 x 32), kept resident.
    v16h af[4];
#pragma unroll
    for (int kk = 0; kk < 4; ++kk) af[kk] = load_frag(fh, r0 + m, kk * 32, hi);

    // Labels of the 8 rows this lane's C-fragment elements belong to.
    int labr[8];
#pragma unroll
    for (int v = 0; v < 8; ++v) labr[v] = labels[r0 + v + 8 * hi];

    float pos[8], all[8];
#pragma unroll
    for (int v = 0; v < 8; ++v) { pos[v] = 0.0f; all[v] = 0.0f; }

    for (int jt = 0; jt < 8; ++jt) {
        const int c0 = cbase + jt * 16;

        v8f c = {};
#pragma unroll
        for (int kk = 0; kk < 4; ++kk) {
            // Gram symmetry: B fragment (K-major in lane) of feats^T column
            // tile == A fragment of those feature rows.
            v16h b = load_frag(fh, c0 + m, kk * 32, hi);
            c = __builtin_amdgcn_wmma_f32_16x16x32_f16(
                    false, af[kk], false, b, (short)0, c, false, false);
        }

        const int col  = c0 + m;           // this lane's column for all 8 rows
        const int labc = labels[col];
#pragma unroll
        for (int v = 0; v < 8; ++v) {
            const int row = r0 + v + 8 * hi;   // C layout: M = v + 8*hi
            const float e = __expf(c[v] * INV_T);
            all[v] += e;
            pos[v] += (labr[v] == labc && row != col) ? e : 0.0f;
        }
    }

    // Row sums live across the 16 lanes of each half-wave -> xor-reduce 1..8.
#pragma unroll
    for (int v = 0; v < 8; ++v) {
        float p = pos[v], a = all[v];
#pragma unroll
        for (int off = 1; off < 16; off <<= 1) {
            p += __shfl_xor(p, off, 32);
            a += __shfl_xor(a, off, 32);
        }
        if (m == 0) {
            const int row = r0 + v + 8 * hi;
            atomicAdd(&Psum[row], p);
            atomicAdd(&Asum[row], a);
        }
    }
}

// ---------------------------------------------------------------------------
// Kernel 3: loss_i = -log(P/(A-P) + 1e-8); mean over rows into d_out[0].
// ---------------------------------------------------------------------------
__global__ void loss_kernel(const float* __restrict__ P,
                            const float* __restrict__ A,
                            float* __restrict__ out) {
    __shared__ float red[8];
    const int i = blockIdx.x * blockDim.x + threadIdx.x;
    float val = 0.0f;
    if (i < NROWS) {
        const float p = P[i];
        const float d = A[i] - p;          // denom: all non-positive + diagonal
        val = -logf(p / d + 1e-8f) * (1.0f / (float)NROWS);
    }
#pragma unroll
    for (int off = 16; off >= 1; off >>= 1) val += __shfl_xor(val, off, 32);
    const int lane = threadIdx.x & 31;
    const int wave = threadIdx.x >> 5;
    if (lane == 0) red[wave] = val;
    __syncthreads();
    if (threadIdx.x == 0) {
        float s = 0.0f;
#pragma unroll
        for (int w = 0; w < 8; ++w) s += red[w];
        atomicAdd(out, s);
    }
}

// ---------------------------------------------------------------------------
extern "C" void kernel_launch(void* const* d_in, const int* in_sizes, int n_in,
                              void* d_out, int out_size, void* d_ws, size_t ws_size,
                              hipStream_t stream) {
    const float* feats  = (const float*)d_in[0];
    const int*   labels = (const int*)d_in[1];
    float*       out    = (float*)d_out;

    char* ws = (char*)d_ws;
    _Float16* fh = (_Float16*)ws;                                   // 4 MB
    float* P = (float*)(ws + (size_t)NROWS * DIM * sizeof(_Float16));
    float* A = P + NROWS;                                           // +128 KB

    zero_kernel<<<(2 * NROWS + 255) / 256, 256, 0, stream>>>(P, out, 2 * NROWS);
    normalize_kernel<<<(NROWS * 32) / 256, 256, 0, stream>>>(feats, fh);

    dim3 grid(NROWS / 128, NROWS / 128);   // 128 x 128 blocks of 128x128 tiles
    supcon_tiles<<<grid, 256, 0, stream>>>(fh, labels, P, A);

    loss_kernel<<<(NROWS + 255) / 256, 256, 0, stream>>>(P, A, out);
}
